// GPTAttention_80719615361630
// MI455X (gfx1250) — matrix operand order repaired and checked by reading, not compiled, not run on hardware
//
#include <hip/hip_runtime.h>
#include <hip/hip_bf16.h>
#include <stdint.h>

// Problem constants (match reference)
#define EMBED   1024
#define HEADS   16
#define HDIM    64
#define BATCH   4
#define SEQ     2048
#define QKVDIM  3072   // 3*EMBED
#define MROWS   8192   // BATCH*SEQ

typedef __attribute__((ext_vector_type(16))) __bf16 v16bf;
typedef __attribute__((ext_vector_type(8)))  float  v8f;
typedef __attribute__((ext_vector_type(4)))  unsigned int u32x4;
typedef __attribute__((ext_vector_type(4)))  __bf16 v4bf;

union FragBF { u32x4 u[2]; v16bf bf; };
union Cvt16  { u32x4 u[2]; __bf16 h[16]; };
union Cvt8   { u32x4 u;    __bf16 h[8]; };

// ---- WMMA bf16 16x16x32, f32 accumulate ----
__device__ inline v8f wmma_bf16(v16bf a, v16bf b, v8f c) {
  return __builtin_amdgcn_wmma_f32_16x16x32_bf16(
      /*neg_a=*/false, a, /*neg_b=*/false, b,
      /*c_mod=*/(short)0, c, /*reuse_a=*/false, /*reuse_b=*/false);
}

// ---- Async global -> LDS copy (CDNA5, ASYNCcnt-tracked) ----
// Per ISA 10.x / 15.18: LDS address is taken from the VDST VGPR; flat addr[31:0] of a
// generic LDS pointer is the LDS byte address, so truncation gives VDST.
__device__ inline void async_copy_b128(void* lds_dst, const void* gsrc) {
  uint32_t l = (uint32_t)(uintptr_t)lds_dst;
  asm volatile("global_load_async_to_lds_b128 %0, %1, off"
               :: "v"(l), "v"(gsrc) : "memory");
}
__device__ inline void wait_async0() {
  asm volatile("s_wait_asynccnt 0" ::: "memory");
}

// A-fragment (16x32 bf16) per ISA layout:
//   lanes 0-15: M=lane, K = kk + half*8 + {0..7} and kk+16+half*8+{0..7}
// Each group of 8 bf16 is one contiguous 16B chunk in LDS -> 2x ds_load_b128.
// rs = row stride in u32x4 (16B) units. kk must be multiple of 8.
__device__ inline v16bf load_a_frag(const u32x4* lds, int mbase, int rs, int kk, int lane) {
  int m = mbase + (lane & 15);
  int half = lane >> 4;
  FragBF f;
  f.u[0] = lds[m * rs + (kk >> 3) + half];
  f.u[1] = lds[m * rs + (kk >> 3) + 2 + half];
  return f.bf;
}

// B-fragment (32x16 bf16) per ISA layout:
//   N = lane&15, K = kk + half*8 + {0..7} and kk+16+half*8+{0..7}
// Each group of 8 bf16 is one contiguous 16B chunk in LDS -> 2x ds_load_b128.
__device__ inline v16bf load_b_frag(const u32x4* lds, int nbase, int rs, int kk, int lane) {
  int n = nbase + (lane & 15);
  int kg = lane >> 4;
  FragBF f;
  f.u[0] = lds[n * rs + (kk >> 3) + kg];
  f.u[1] = lds[n * rs + (kk >> 3) + 2 + kg];
  return f.bf;
}

// ---------------------------------------------------------------------------
// K0: elementwise f32 -> bf16 (vectorized x4)
// ---------------------------------------------------------------------------
__global__ __launch_bounds__(256) void cvt_f32_bf16_kernel(
    const float* __restrict__ in, __bf16* __restrict__ out, int n4) {
  int gid = blockIdx.x * 256 + threadIdx.x;
  if (gid >= n4) return;
  float4 v = reinterpret_cast<const float4*>(in)[gid];
  v4bf o;
  o[0] = (__bf16)v.x; o[1] = (__bf16)v.y; o[2] = (__bf16)v.z; o[3] = (__bf16)v.w;
  reinterpret_cast<v4bf*>(out)[gid] = o;
}

// ---------------------------------------------------------------------------
// K1/K5: C[M,N] = A[M,K] * B[N,K]^T + bias[n], NT bf16 GEMM.
// Block 128x128x32, 256 threads = 8 waves as 2x4; wave tile 64x32 (4x2 WMMA).
// Double-buffered async global->LDS pipeline: compute tile t from LDS buf t%2
// while the async engine fills buf (t+1)%2.
// ---------------------------------------------------------------------------
template <bool OUT_BF16>
__global__ __launch_bounds__(256) void gemm_nt_kernel(
    const __bf16* __restrict__ A, const __bf16* __restrict__ Bw,
    const float* __restrict__ bias, void* __restrict__ Cp,
    int M, int N, int K, int scale_cols, float scale_val) {
  __shared__ alignas(16) __bf16 As[2][128 * 32];
  __shared__ alignas(16) __bf16 Bs[2][128 * 32];
  const int tid = threadIdx.x, lane = tid & 31, wave = tid >> 5;
  const int wm = wave >> 2, wn = wave & 3;
  const int bm0 = blockIdx.x * 128, bn0 = blockIdx.y * 128;

  v8f acc[4][2];
  for (int i = 0; i < 4; ++i)
    for (int j = 0; j < 2; ++j)
      for (int r = 0; r < 8; ++r) acc[i][j][r] = 0.0f;

  // per-thread tile slice: 2 chunks of 8 bf16 for A, same for B
  const int r0 = tid >> 2;               // rows 0..63  (chunk 0)
  const int r1 = 64 + (tid >> 2);        // rows 64..127 (chunk 1)
  const int kc = (tid & 3) * 8;          // 0,8,16,24

  auto issue_tile = [&](int k0, int buf) {
    async_copy_b128(&As[buf][r0 * 32 + kc], &A[(size_t)(bm0 + r0) * K + k0 + kc]);
    async_copy_b128(&As[buf][r1 * 32 + kc], &A[(size_t)(bm0 + r1) * K + k0 + kc]);
    async_copy_b128(&Bs[buf][r0 * 32 + kc], &Bw[(size_t)(bn0 + r0) * K + k0 + kc]);
    async_copy_b128(&Bs[buf][r1 * 32 + kc], &Bw[(size_t)(bn0 + r1) * K + k0 + kc]);
  };

  const int nk = K >> 5;
  issue_tile(0, 0);
  for (int t = 0; t < nk; ++t) {
    wait_async0();
    __syncthreads();   // buf t%2 fully written; buf (t+1)%2 fully consumed
    if (t + 1 < nk) issue_tile((t + 1) << 5, (t + 1) & 1);

    const u32x4* Au = reinterpret_cast<const u32x4*>(As[t & 1]);
    const u32x4* Bu = reinterpret_cast<const u32x4*>(Bs[t & 1]);
    v16bf af[4], bf[2];
#pragma unroll
    for (int i = 0; i < 4; ++i) af[i] = load_a_frag(Au, wm * 64 + i * 16, 4, 0, lane);
#pragma unroll
    for (int j = 0; j < 2; ++j) bf[j] = load_b_frag(Bu, wn * 32 + j * 16, 4, 0, lane);
#pragma unroll
    for (int i = 0; i < 4; ++i)
#pragma unroll
      for (int j = 0; j < 2; ++j) acc[i][j] = wmma_bf16(af[i], bf[j], acc[i][j]);
  }

  // Epilogue. C/D layout: col = lane&15 within 16-tile; M = vgpr + (lane>>4)*8.
#pragma unroll
  for (int i = 0; i < 4; ++i) {
    int row0 = bm0 + wm * 64 + i * 16 + ((lane >> 4) << 3);
#pragma unroll
    for (int j = 0; j < 2; ++j) {
      int col = bn0 + wn * 32 + j * 16 + (lane & 15);
      float bv = bias[col];
      float sc = (col < scale_cols) ? scale_val : 1.0f;
#pragma unroll
      for (int r = 0; r < 8; ++r) {
        float v = (acc[i][j][r] + bv) * sc;
        size_t off = (size_t)(row0 + r) * N + col;
        if (OUT_BF16) ((__bf16*)Cp)[off] = (__bf16)v;
        else          ((float*)Cp)[off] = v;
      }
    }
  }
}

// ---------------------------------------------------------------------------
// K2: scores[b,h,q,k] = Qs . K^T, causal masked, f32 to d_out attn region.
// Q already pre-scaled by 1/sqrt(D) in K1 epilogue. K-dim = HDIM = 64.
// Q/K tiles staged via async global->LDS copies.
// ---------------------------------------------------------------------------
__global__ __launch_bounds__(256) void attn_scores_kernel(
    const __bf16* __restrict__ qkv, float* __restrict__ attnW) {
  const int bh = blockIdx.z;
  const int b = bh >> 4, h = bh & 15;
  const int bm0 = blockIdx.x * 128;  // query tile
  const int bn0 = blockIdx.y * 128;  // key tile
  float* Crow = attnW + (size_t)bh * SEQ * SEQ;
  const int tid = threadIdx.x;

  if (bn0 > bm0 + 127) {
    // Fully above the diagonal: masked; softmax turns -1e30 into 0.
#pragma unroll
    for (int c = 0; c < 16; ++c) {
      int idx = c * 1024 + tid * 4;
      int r = idx >> 7, col = idx & 127;
      float4 m4 = make_float4(-1e30f, -1e30f, -1e30f, -1e30f);
      *reinterpret_cast<float4*>(&Crow[(size_t)(bm0 + r) * SEQ + bn0 + col]) = m4;
    }
    return;
  }

  __shared__ alignas(16) __bf16 Qs[128 * 64];
  __shared__ alignas(16) __bf16 Ks[128 * 64];
  const int lane = tid & 31, wave = tid >> 5;
  const int wm = wave >> 2, wn = wave & 3;
  const __bf16* Qb = qkv + (size_t)b * SEQ * QKVDIM + h * HDIM;  // Q slice
  const __bf16* Kb = Qb + EMBED;                                 // K slice

#pragma unroll
  for (int c = 0; c < 4; ++c) {
    int idx = c * 2048 + tid * 8;
    int r = idx >> 6, kc = idx & 63;
    async_copy_b128(&Qs[r * 64 + kc], &Qb[(size_t)(bm0 + r) * QKVDIM + kc]);
    async_copy_b128(&Ks[r * 64 + kc], &Kb[(size_t)(bn0 + r) * QKVDIM + kc]);
  }
  wait_async0();
  __syncthreads();

  v8f acc[4][2];
  for (int i = 0; i < 4; ++i)
    for (int j = 0; j < 2; ++j)
      for (int r = 0; r < 8; ++r) acc[i][j][r] = 0.0f;

  const u32x4* Qu = reinterpret_cast<const u32x4*>(Qs);
  const u32x4* Ku = reinterpret_cast<const u32x4*>(Ks);
#pragma unroll
  for (int kk = 0; kk < 64; kk += 32) {
    v16bf af[4], bf[2];
#pragma unroll
    for (int i = 0; i < 4; ++i) af[i] = load_a_frag(Qu, wm * 64 + i * 16, 8, kk, lane);
#pragma unroll
    for (int j = 0; j < 2; ++j) bf[j] = load_b_frag(Ku, wn * 32 + j * 16, 8, kk, lane);
#pragma unroll
    for (int i = 0; i < 4; ++i)
#pragma unroll
      for (int j = 0; j < 2; ++j) acc[i][j] = wmma_bf16(af[i], bf[j], acc[i][j]);
  }

#pragma unroll
  for (int i = 0; i < 4; ++i) {
    int row0 = bm0 + wm * 64 + i * 16 + ((lane >> 4) << 3);
#pragma unroll
    for (int j = 0; j < 2; ++j) {
      int col = bn0 + wn * 32 + j * 16 + (lane & 15);
#pragma unroll
      for (int r = 0; r < 8; ++r) {
        int row = row0 + r;
        float v = (col > row) ? -1e30f : acc[i][j][r];
        Crow[(size_t)row * SEQ + col] = v;
      }
    }
  }
}

// ---------------------------------------------------------------------------
// K3: in-place row softmax over S=2048 f32 per block (256 thr, 8 vals/thr).
// ---------------------------------------------------------------------------
__global__ __launch_bounds__(256) void softmax_kernel(float* __restrict__ attnW) {
  __shared__ float red[256];
  float* p = attnW + (size_t)blockIdx.x * SEQ;
  const int tid = threadIdx.x;
  float4 v0 = reinterpret_cast<float4*>(p)[tid * 2];
  float4 v1 = reinterpret_cast<float4*>(p)[tid * 2 + 1];
  float m = fmaxf(fmaxf(fmaxf(v0.x, v0.y), fmaxf(v0.z, v0.w)),
                  fmaxf(fmaxf(v1.x, v1.y), fmaxf(v1.z, v1.w)));
  red[tid] = m;
  __syncthreads();
  for (int s = 128; s > 0; s >>= 1) {
    if (tid < s) red[tid] = fmaxf(red[tid], red[tid + s]);
    __syncthreads();
  }
  m = red[0];
  __syncthreads();
  v0.x = __expf(v0.x - m); v0.y = __expf(v0.y - m);
  v0.z = __expf(v0.z - m); v0.w = __expf(v0.w - m);
  v1.x = __expf(v1.x - m); v1.y = __expf(v1.y - m);
  v1.z = __expf(v1.z - m); v1.w = __expf(v1.w - m);
  float sum = v0.x + v0.y + v0.z + v0.w + v1.x + v1.y + v1.z + v1.w;
  red[tid] = sum;
  __syncthreads();
  for (int s = 128; s > 0; s >>= 1) {
    if (tid < s) red[tid] += red[tid + s];
    __syncthreads();
  }
  float inv = 1.0f / red[0];
  v0.x *= inv; v0.y *= inv; v0.z *= inv; v0.w *= inv;
  v1.x *= inv; v1.y *= inv; v1.z *= inv; v1.w *= inv;
  reinterpret_cast<float4*>(p)[tid * 2] = v0;
  reinterpret_cast<float4*>(p)[tid * 2 + 1] = v1;
}

// ---------------------------------------------------------------------------
// K4: attn_out[b,q,h*64+d] = sum_k P[b,h,q,k] * V[b,k,h*64+d]  (bf16 out)
// Block 128x64x32; 8 waves as 4x2; wave tile 32x32 (2x2 WMMA).
// P converted f32->bf16 in registers while staging; V transposed into LDS.
// ---------------------------------------------------------------------------
__global__ __launch_bounds__(256) void attn_pv_kernel(
    const float* __restrict__ attnW, const __bf16* __restrict__ qkv,
    __bf16* __restrict__ ao) {
  const int bh = blockIdx.y;
  const int b = bh >> 4, h = bh & 15;
  const int bm0 = blockIdx.x * 128;
  __shared__ alignas(16) __bf16 Ps[128 * 32];
  __shared__ alignas(16) __bf16 Vt[64 * 32];
  const int tid = threadIdx.x, lane = tid & 31, wave = tid >> 5;
  const int wm = wave >> 1, wn = wave & 1;
  const float* Pb = attnW + (size_t)bh * SEQ * SEQ;
  const __bf16* Vb = qkv + (size_t)b * SEQ * QKVDIM + 2 * EMBED + h * HDIM;

  v8f acc[2][2];
  for (int i = 0; i < 2; ++i)
    for (int j = 0; j < 2; ++j)
      for (int r = 0; r < 8; ++r) acc[i][j][r] = 0.0f;

  for (int k0 = 0; k0 < SEQ; k0 += 32) {
    {  // stage P tile 128x32, f32 -> bf16
      int r = tid >> 1, half = tid & 1;
      const float4* s4 = reinterpret_cast<const float4*>(
          &Pb[(size_t)(bm0 + r) * SEQ + k0 + half * 16]);
      float4 f0 = s4[0], f1 = s4[1], f2 = s4[2], f3 = s4[3];
      if (k0 + 32 < SEQ)  // global_prefetch_b8 of next P tile
        __builtin_prefetch(&Pb[(size_t)(bm0 + r) * SEQ + k0 + 32 + half * 16], 0, 1);
      Cvt16 cv;
      cv.h[0] = (__bf16)f0.x;  cv.h[1] = (__bf16)f0.y;
      cv.h[2] = (__bf16)f0.z;  cv.h[3] = (__bf16)f0.w;
      cv.h[4] = (__bf16)f1.x;  cv.h[5] = (__bf16)f1.y;
      cv.h[6] = (__bf16)f1.z;  cv.h[7] = (__bf16)f1.w;
      cv.h[8] = (__bf16)f2.x;  cv.h[9] = (__bf16)f2.y;
      cv.h[10] = (__bf16)f2.z; cv.h[11] = (__bf16)f2.w;
      cv.h[12] = (__bf16)f3.x; cv.h[13] = (__bf16)f3.y;
      cv.h[14] = (__bf16)f3.z; cv.h[15] = (__bf16)f3.w;
      *reinterpret_cast<u32x4*>(&Ps[r * 32 + half * 16]) = cv.u[0];
      *reinterpret_cast<u32x4*>(&Ps[r * 32 + half * 16 + 8]) = cv.u[1];
    }
    {  // stage V tile 32(k)x64(n), transposed into Vt[n][k]
      int idx = tid * 8;
      int k = idx >> 6, n0 = idx & 63;
      Cvt8 cv;
      cv.u = *reinterpret_cast<const u32x4*>(&Vb[(size_t)(k0 + k) * QKVDIM + n0]);
#pragma unroll
      for (int e = 0; e < 8; ++e) Vt[(n0 + e) * 32 + k] = cv.h[e];
    }
    __syncthreads();
    const u32x4* Pu = reinterpret_cast<const u32x4*>(Ps);
    const u32x4* Vu = reinterpret_cast<const u32x4*>(Vt);
    v16bf af[2], bf[2];
#pragma unroll
    for (int i = 0; i < 2; ++i) af[i] = load_a_frag(Pu, wm * 32 + i * 16, 4, 0, lane);
#pragma unroll
    for (int j = 0; j < 2; ++j) bf[j] = load_b_frag(Vu, wn * 32 + j * 16, 4, 0, lane);
#pragma unroll
    for (int i = 0; i < 2; ++i)
#pragma unroll
      for (int j = 0; j < 2; ++j) acc[i][j] = wmma_bf16(af[i], bf[j], acc[i][j]);
    __syncthreads();
  }

#pragma unroll
  for (int i = 0; i < 2; ++i) {
    int row0 = bm0 + wm * 32 + i * 16 + ((lane >> 4) << 3);
#pragma unroll
    for (int j = 0; j < 2; ++j) {
      int col = wn * 32 + j * 16 + (lane & 15);
#pragma unroll
      for (int r = 0; r < 8; ++r) {
        ao[(size_t)(b * SEQ + row0 + r) * EMBED + h * HDIM + col] =
            (__bf16)acc[i][j][r];
      }
    }
  }
}

// ---------------------------------------------------------------------------
extern "C" void kernel_launch(void* const* d_in, const int* in_sizes, int n_in,
                              void* d_out, int out_size, void* d_ws, size_t ws_size,
                              hipStream_t stream) {
  const float* x     = (const float*)d_in[0];
  const float* qkv_w = (const float*)d_in[1];
  const float* qkv_b = (const float*)d_in[2];
  const float* out_w = (const float*)d_in[3];
  const float* out_b = (const float*)d_in[4];

  float* outp  = (float*)d_out;
  float* attnW = outp + (size_t)BATCH * SEQ * EMBED;  // (B,H,S,S) f32

  // Workspace layout (bf16): x | qkv_w | out_w | qkv | attn_out  (~88 MB)
  char* ws = (char*)d_ws;
  __bf16* xb   = (__bf16*)ws;
  __bf16* wqkv = xb + (size_t)BATCH * SEQ * EMBED;
  __bf16* wout = wqkv + (size_t)QKVDIM * EMBED;
  __bf16* qkv  = wout + (size_t)EMBED * EMBED;
  __bf16* ao   = qkv + (size_t)BATCH * SEQ * QKVDIM;

  // K0: precision conversions
  {
    int n4 = BATCH * SEQ * EMBED / 4;
    cvt_f32_bf16_kernel<<<(n4 + 255) / 256, 256, 0, stream>>>(x, xb, n4);
    n4 = QKVDIM * EMBED / 4;
    cvt_f32_bf16_kernel<<<(n4 + 255) / 256, 256, 0, stream>>>(qkv_w, wqkv, n4);
    n4 = EMBED * EMBED / 4;
    cvt_f32_bf16_kernel<<<(n4 + 255) / 256, 256, 0, stream>>>(out_w, wout, n4);
  }

  // K1: QKV projection (Q columns pre-scaled by 1/sqrt(64) = 0.125)
  gemm_nt_kernel<true><<<dim3(MROWS / 128, QKVDIM / 128), 256, 0, stream>>>(
      xb, wqkv, qkv_b, (void*)qkv, MROWS, QKVDIM, EMBED, EMBED, 0.125f);

  // K2: causal-masked scores -> d_out attn region
  attn_scores_kernel<<<dim3(SEQ / 128, SEQ / 128, BATCH * HEADS), 256, 0, stream>>>(
      qkv, attnW);

  // K3: softmax rows in place
  softmax_kernel<<<BATCH * HEADS * SEQ, 256, 0, stream>>>(attnW);

  // K4: P @ V -> attn_out (bf16, already (B,S,E) layout)
  attn_pv_kernel<<<dim3(SEQ / 128, BATCH * HEADS), 256, 0, stream>>>(attnW, qkv, ao);

  // K5: output projection + bias -> f32 d_out
  gemm_nt_kernel<false><<<dim3(MROWS / 128, EMBED / 128), 256, 0, stream>>>(
      ao, wout, out_b, (void*)outp, MROWS, EMBED, EMBED, 0, 1.0f);
}
